// CohomologicalAttention_51642686767400
// MI455X (gfx1250) — matrix operand order repaired and checked by reading, not compiled, not run on hardware
//
#include <hip/hip_runtime.h>

// ---------------------------------------------------------------------------
// CohomologicalAttention on MI455X (gfx1250, wave32, WMMA bf16 + TDM)
// cw == softmax over size-1 axis == 1.0 identically -> Wr/br path is dead.
// Remaining math: standard MHA. B=2,S=2048,E=1024,H=16,DH=64.
// ---------------------------------------------------------------------------

typedef __bf16 bf16;
typedef __attribute__((ext_vector_type(16))) __bf16 v16bf;
typedef __attribute__((ext_vector_type(8)))  __bf16 v8bf;
typedef __attribute__((ext_vector_type(8)))  float  v8f;
typedef __attribute__((ext_vector_type(4)))  unsigned int u32x4;
typedef __attribute__((ext_vector_type(8)))  int i32x8;
typedef __attribute__((ext_vector_type(4)))  int i32x4;

#define B_ 2
#define S_ 2048
#define E_ 1024
#define H_ 16
#define DH_ 64
#define M_TOTAL (B_ * S_)       // 4096 rows of x
#define KCHUNK 128              // K panel staged in LDS per TDM transfer
#define NCHUNK (E_ / KCHUNK)    // 8

union ABfrag { v16bf v; v8bf h[2]; };

__device__ inline v8f vzero8() {
  v8f z = {0.f, 0.f, 0.f, 0.f, 0.f, 0.f, 0.f, 0.f};
  return z;
}

__device__ inline v8f wmma_bf16(v16bf a, v16bf b, v8f c) {
  // 8 args: (neg_a, A, neg_b, B, c_mod, C, reuse_a, reuse_b)
  return __builtin_amdgcn_wmma_f32_16x16x32_bf16(false, a, false, b, (short)0, c,
                                                 false, false);
}

// A fragment (16x32, MxK) from row-major A[m][k] (global or LDS).
// lane m = m0 + (lane&15); sel=lane>>4;
//   VGPR0..3: K = k0 + sel*8 + 0..7 ; VGPR4..7: K = k0 + 16 + sel*8 + 0..7
__device__ inline v16bf load_a_frag(const bf16* __restrict__ A, int lda,
                                    int m0, int k0, int lane) {
  const bf16* p = A + (size_t)(m0 + (lane & 15)) * lda + k0 + ((lane >> 4) << 3);
  ABfrag f;
  f.h[0] = *(const v8bf*)(p);
  f.h[1] = *(const v8bf*)(p + 16);
  return f.v;
}

// B fragment (32x16, KxN) from "n-major" storage Wn[n][k] (B[k][n] = Wn[n][k]).
// lane n = n0 + (lane&15); K = k0 + (lane>>4)*8 + 0..7, then k0 + 16 + (lane>>4)*8 + 0..7.
__device__ inline v16bf load_b_frag_nmajor(const bf16* __restrict__ Wn, int ldk,
                                           int n0, int k0, int lane) {
  const bf16* p = Wn + (size_t)(n0 + (lane & 15)) * ldk + k0 + ((lane >> 4) << 3);
  ABfrag f;
  f.h[0] = *(const v8bf*)(p);
  f.h[1] = *(const v8bf*)(p + 16);
  return f.v;
}

// ---------------------------------------------------------------------------
// TDM: stage a [tile_d1=32 rows x tile_d0=128 cols] bf16 tile into LDS.
// D# per CDNA5 ISA (2D: groups 2/3 zero).
// This toolchain exposes the 6-arg builtin:
//   (u32x4 g0, i32x8 g1, i32x4 g2, i32x4 g3, i32x8 extra, i32 cpol)
// ---------------------------------------------------------------------------
#if __has_builtin(__builtin_amdgcn_tensor_load_to_lds)
#define HAVE_TDM 1
__device__ inline void tdm_load_tile_2d(const bf16* gsrc, unsigned lds_byte,
                                        unsigned tensor_d0, unsigned tensor_d1,
                                        unsigned stride_d0, unsigned tile_d0,
                                        unsigned tile_d1) {
  unsigned long long ga = (unsigned long long)(uintptr_t)gsrc;
  u32x4 g0;
  g0[0] = 1u;                                     // count=1, user descriptor
  g0[1] = lds_byte;                               // lds_addr [63:32]
  g0[2] = (unsigned)(ga & 0xffffffffu);           // global_addr lo
  g0[3] = (unsigned)((ga >> 32) & 0x01ffffffu)    // global_addr [56:32]
          | 0x80000000u;                          // type=2 ("image")
  i32x8 g1;
  g1[0] = (int)(1u << 16);                        // data_size=1 (2B); mask=0
  g1[1] = (int)((tensor_d0 & 0xffffu) << 16);     // tensor_dim0[15:0] @ [31:16]
  g1[2] = (int)(((tensor_d0 >> 16) & 0xffffu) |
                ((tensor_d1 & 0xffffu) << 16));   // dim0 hi | dim1 lo
  g1[3] = (int)(((tensor_d1 >> 16) & 0xffffu) |
                ((tile_d0 & 0xffffu) << 16));     // dim1 hi | tile_dim0
  g1[4] = (int)(tile_d1 & 0xffffu);               // tile_dim1 ; tile_dim2=0
  g1[5] = (int)stride_d0;                         // tensor_dim0_stride[31:0]
  g1[6] = 0;                                      // stride0 hi | stride1 lo
  g1[7] = 0;                                      // stride1 hi
  i32x4 z4 = {0, 0, 0, 0};
  i32x8 z8 = {0, 0, 0, 0, 0, 0, 0, 0};
  __builtin_amdgcn_tensor_load_to_lds(g0, g1, z4, z4, z8, 0);
}
#else
#define HAVE_TDM 0
#endif

// Stage one 32 x KCHUNK A-panel chunk into ldsbuf (row-major, ld=KCHUNK).
__device__ inline void stage_a_chunk(const bf16* __restrict__ gtile,
                                     bf16* __restrict__ ldsbuf, int src_ld,
                                     bool issuer_wave, int tid) {
#if HAVE_TDM
  (void)tid;
  if (issuer_wave) {
    tdm_load_tile_2d(gtile, (unsigned)(unsigned long long)(uintptr_t)ldsbuf,
                     (unsigned)src_ld, (unsigned)M_TOTAL, (unsigned)src_ld,
                     KCHUNK, 32);
  }
#else
  (void)issuer_wave;
  for (int idx = tid; idx < 32 * (KCHUNK / 8); idx += 128) {
    const int row = idx >> 4;          // KCHUNK/8 == 16 chunks per row
    const int col = (idx & 15) * 8;
    *(v8bf*)(ldsbuf + row * KCHUNK + col) =
        *(const v8bf*)(gtile + (size_t)row * src_ld + col);
  }
#endif
}

__device__ inline void stage_wait(bool issuer_wave) {
#if HAVE_TDM
  if (issuer_wave) __builtin_amdgcn_s_wait_tensorcnt(0);
#else
  (void)issuer_wave;
#endif
}

// ---------------------------------------------------------------------------
// fp32 -> bf16 convert (grid-stride)
// ---------------------------------------------------------------------------
__global__ void cvt_bf16_kernel(const float* __restrict__ src,
                                bf16* __restrict__ dst, int n) {
  int i = blockIdx.x * blockDim.x + threadIdx.x;
  int stride = gridDim.x * blockDim.x;
  for (; i < n; i += stride) dst[i] = (bf16)src[i];
}

__global__ void fill_ones_kernel(float* __restrict__ p, int n) {
  int i = blockIdx.x * blockDim.x + threadIdx.x;
  if (i < n) p[i] = 1.0f;
}

// ---------------------------------------------------------------------------
// Fused QKV projection: out = x @ W.T + b, split into heads.
// Block tile M=32 x N=256; 4 waves along N (64 cols each) share the A panel,
// which is TDM-staged into double-buffered LDS and overlapped with compute.
// Q,K written [B,H,S,DH]; V written transposed [B,H,DH,S] for the PV GEMM.
// Grid: (M/32, N/256, 3).
// ---------------------------------------------------------------------------
__global__ __launch_bounds__(128) void qkv_gemm_kernel(
    const bf16* __restrict__ xb,
    const bf16* __restrict__ wq, const bf16* __restrict__ wk,
    const bf16* __restrict__ wv,
    const float* __restrict__ bq, const float* __restrict__ bk,
    const float* __restrict__ bv,
    bf16* __restrict__ qout, bf16* __restrict__ kout, bf16* __restrict__ vTout) {
  __shared__ __align__(16) bf16 lds_a[2][32 * KCHUNK];  // 16 KB double buffer

  const int lane = threadIdx.x & 31;
  const int wave = threadIdx.x >> 5;
  const int m0 = blockIdx.x * 32;
  const int n0 = blockIdx.y * 256 + wave * 64;
  const int which = blockIdx.z;
  const bool issuer = (wave == 0);

  const bf16* W = (which == 0) ? wq : (which == 1) ? wk : wv;
  const float* bias = (which == 0) ? bq : (which == 1) ? bk : bv;
  const bf16* atile = xb + (size_t)m0 * E_;

  v8f acc[2][4];
#pragma unroll
  for (int i = 0; i < 2; ++i)
#pragma unroll
    for (int j = 0; j < 4; ++j) acc[i][j] = vzero8();

  // prologue: stage chunk 0, wait, release block
  stage_a_chunk(atile, &lds_a[0][0], E_, issuer, threadIdx.x);
  stage_wait(issuer);
  __syncthreads();

  for (int c = 0; c < NCHUNK; ++c) {
    // kick DMA for chunk c+1 while computing on chunk c
    if (c + 1 < NCHUNK) {
      stage_a_chunk(atile + (c + 1) * KCHUNK, &lds_a[(c + 1) & 1][0], E_,
                    issuer, threadIdx.x);
      __builtin_prefetch(W + (size_t)(n0 + (lane & 15)) * E_ + (c + 1) * KCHUNK,
                         0, 1);
    }
    const bf16* abuf = &lds_a[c & 1][0];
#pragma unroll
    for (int kk = 0; kk < KCHUNK; kk += 32) {
      const v16bf a0 = load_a_frag(abuf, KCHUNK, 0, kk, lane);   // rows 0..15
      const v16bf a1 = load_a_frag(abuf, KCHUNK, 16, kk, lane);  // rows 16..31
      const int k0 = c * KCHUNK + kk;
#pragma unroll
      for (int nt = 0; nt < 4; ++nt) {
        const v16bf b = load_b_frag_nmajor(W, E_, n0 + nt * 16, k0, lane);
        acc[0][nt] = wmma_bf16(a0, b, acc[0][nt]);
        acc[1][nt] = wmma_bf16(a1, b, acc[1][nt]);
      }
    }
    if (c + 1 < NCHUNK) stage_wait(issuer);
    __syncthreads();
  }

  const int sel = lane >> 4;
  const int cn = lane & 15;
#pragma unroll
  for (int mh = 0; mh < 2; ++mh) {
#pragma unroll
    for (int nt = 0; nt < 4; ++nt) {
      const int n = n0 + nt * 16 + cn;
      const int h = n >> 6;          // n / 64
      const int d = n & 63;
      const float bval = bias[n];
#pragma unroll
      for (int r = 0; r < 8; ++r) {
        const int m = m0 + mh * 16 + r + 8 * sel;
        const int bb = m >> 11;      // m / S_
        const int s = m & (S_ - 1);
        const bf16 v = (bf16)(acc[mh][nt][r] + bval);
        const size_t bh = (size_t)(bb * H_ + h);
        if (which == 0)      qout[(bh * S_ + s) * DH_ + d] = v;
        else if (which == 1) kout[(bh * S_ + s) * DH_ + d] = v;
        else                 vTout[(bh * DH_ + d) * S_ + s] = v;
      }
    }
  }
}

// ---------------------------------------------------------------------------
// Flash attention per (b,h): each wave owns a 16-query tile, streams keys in
// 32-wide steps. 8 WMMAs per step; online softmax with 16-lane butterflies;
// P tile transposed to A-fragment layout through LDS.
// Grid: (S/64, B*H). Block: 128 threads = 4 waves.
// ---------------------------------------------------------------------------
__global__ __launch_bounds__(128) void attn_kernel(
    const bf16* __restrict__ qten, const bf16* __restrict__ kten,
    const bf16* __restrict__ vTten, bf16* __restrict__ attnb) {
  __shared__ __align__(16) bf16 lds_p[4][16 * 32];

  const int lane = threadIdx.x & 31;
  const int wave = threadIdx.x >> 5;
  const int bh = blockIdx.y;              // 0..31
  const int b = bh >> 4;
  const int h = bh & 15;
  const int s0 = (blockIdx.x * 4 + wave) * 16;

  const bf16* Q  = qten  + (size_t)bh * S_ * DH_;
  const bf16* K  = kten  + (size_t)bh * S_ * DH_;
  const bf16* VT = vTten + (size_t)bh * DH_ * S_;

  const v16bf q0 = load_a_frag(Q, DH_, s0, 0, lane);
  const v16bf q1 = load_a_frag(Q, DH_, s0, 32, lane);

  v8f o[4];
#pragma unroll
  for (int i = 0; i < 4; ++i) o[i] = vzero8();
  float mrow[8], lrow[8];
#pragma unroll
  for (int r = 0; r < 8; ++r) { mrow[r] = -1.0e30f; lrow[r] = 0.0f; }

  const int sel = lane >> 4;
  const int cn = lane & 15;
  bf16* lp = &lds_p[wave][0];

  for (int t0 = 0; t0 < S_; t0 += 32) {
    // scores tile: Q(16x64) x K^T(64x32) -> two 16x16 f32 tiles
    v8f sc0 = vzero8(), sc1 = vzero8();
    sc0 = wmma_bf16(q0, load_b_frag_nmajor(K, DH_, t0, 0, lane), sc0);
    sc0 = wmma_bf16(q1, load_b_frag_nmajor(K, DH_, t0, 32, lane), sc0);
    sc1 = wmma_bf16(q0, load_b_frag_nmajor(K, DH_, t0 + 16, 0, lane), sc1);
    sc1 = wmma_bf16(q1, load_b_frag_nmajor(K, DH_, t0 + 16, 32, lane), sc1);

#pragma unroll
    for (int r = 0; r < 8; ++r) {
      const float v0 = sc0[r] * 0.125f;     // 1/sqrt(64)
      const float v1 = sc1[r] * 0.125f;
      float tmax = fmaxf(v0, v1);
      tmax = fmaxf(tmax, __shfl_xor(tmax, 1));
      tmax = fmaxf(tmax, __shfl_xor(tmax, 2));
      tmax = fmaxf(tmax, __shfl_xor(tmax, 4));
      tmax = fmaxf(tmax, __shfl_xor(tmax, 8));
      const float mnew = fmaxf(mrow[r], tmax);
      const float alpha = __expf(mrow[r] - mnew);
      const float p0 = __expf(v0 - mnew);
      const float p1 = __expf(v1 - mnew);
      float tsum = p0 + p1;
      tsum += __shfl_xor(tsum, 1);
      tsum += __shfl_xor(tsum, 2);
      tsum += __shfl_xor(tsum, 4);
      tsum += __shfl_xor(tsum, 8);
      lrow[r] = lrow[r] * alpha + tsum;
      mrow[r] = mnew;
      o[0][r] *= alpha; o[1][r] *= alpha; o[2][r] *= alpha; o[3][r] *= alpha;
      // stage P (C-layout -> LDS row-major [16][32])
      const int row = r + 8 * sel;
      lp[row * 32 + cn]      = (bf16)p0;
      lp[row * 32 + 16 + cn] = (bf16)p1;
    }

    // reload P as an A fragment (16x32), then O += P x V(32x64)
    const bf16* pp = lp + cn * 32 + (sel << 3);
    ABfrag pf;
    pf.h[0] = *(const v8bf*)(pp);
    pf.h[1] = *(const v8bf*)(pp + 16);
#pragma unroll
    for (int dt = 0; dt < 4; ++dt) {
      v16bf bv = load_b_frag_nmajor(VT, S_, dt * 16, t0, lane);
      o[dt] = wmma_bf16(pf.v, bv, o[dt]);
    }
  }

  // normalize and write attended [B,S,E] in bf16 (col = h*64 + d)
#pragma unroll
  for (int r = 0; r < 8; ++r) {
    const float inv = 1.0f / lrow[r];
    const int srow = s0 + r + 8 * sel;
    const size_t rowbase = ((size_t)b * S_ + srow) * E_ + h * DH_;
#pragma unroll
    for (int dt = 0; dt < 4; ++dt) {
      attnb[rowbase + dt * 16 + cn] = (bf16)(o[dt][r] * inv);
    }
  }
}

// ---------------------------------------------------------------------------
// Output projection: out = attended @ Wo.T + bo (fp32 result).
// Same TDM-staged structure as the QKV GEMM. Grid: (M/32, N/256).
// ---------------------------------------------------------------------------
__global__ __launch_bounds__(128) void out_gemm_kernel(
    const bf16* __restrict__ attnb, const bf16* __restrict__ wo,
    const float* __restrict__ bo, float* __restrict__ out) {
  __shared__ __align__(16) bf16 lds_a[2][32 * KCHUNK];

  const int lane = threadIdx.x & 31;
  const int wave = threadIdx.x >> 5;
  const int m0 = blockIdx.x * 32;
  const int n0 = blockIdx.y * 256 + wave * 64;
  const bool issuer = (wave == 0);
  const bf16* atile = attnb + (size_t)m0 * E_;

  v8f acc[2][4];
#pragma unroll
  for (int i = 0; i < 2; ++i)
#pragma unroll
    for (int j = 0; j < 4; ++j) acc[i][j] = vzero8();

  stage_a_chunk(atile, &lds_a[0][0], E_, issuer, threadIdx.x);
  stage_wait(issuer);
  __syncthreads();

  for (int c = 0; c < NCHUNK; ++c) {
    if (c + 1 < NCHUNK) {
      stage_a_chunk(atile + (c + 1) * KCHUNK, &lds_a[(c + 1) & 1][0], E_,
                    issuer, threadIdx.x);
      __builtin_prefetch(wo + (size_t)(n0 + (lane & 15)) * E_ + (c + 1) * KCHUNK,
                         0, 1);
    }
    const bf16* abuf = &lds_a[c & 1][0];
#pragma unroll
    for (int kk = 0; kk < KCHUNK; kk += 32) {
      const v16bf a0 = load_a_frag(abuf, KCHUNK, 0, kk, lane);
      const v16bf a1 = load_a_frag(abuf, KCHUNK, 16, kk, lane);
      const int k0 = c * KCHUNK + kk;
#pragma unroll
      for (int nt = 0; nt < 4; ++nt) {
        const v16bf b = load_b_frag_nmajor(wo, E_, n0 + nt * 16, k0, lane);
        acc[0][nt] = wmma_bf16(a0, b, acc[0][nt]);
        acc[1][nt] = wmma_bf16(a1, b, acc[1][nt]);
      }
    }
    if (c + 1 < NCHUNK) stage_wait(issuer);
    __syncthreads();
  }

  const int sel = lane >> 4;
  const int cn = lane & 15;
#pragma unroll
  for (int mh = 0; mh < 2; ++mh) {
#pragma unroll
    for (int nt = 0; nt < 4; ++nt) {
      const int n = n0 + nt * 16 + cn;
      const float bval = bo[n];
#pragma unroll
      for (int r = 0; r < 8; ++r) {
        const int m = m0 + mh * 16 + r + 8 * sel;
        out[(size_t)m * E_ + n] = acc[mh][nt][r] + bval;
      }
    }
  }
}

// ---------------------------------------------------------------------------
// Host-side launch
// ---------------------------------------------------------------------------
extern "C" void kernel_launch(void* const* d_in, const int* in_sizes, int n_in,
                              void* d_out, int out_size, void* d_ws, size_t ws_size,
                              hipStream_t stream) {
  const float* x  = (const float*)d_in[0];
  const float* Wq = (const float*)d_in[1];
  const float* bq = (const float*)d_in[2];
  const float* Wk = (const float*)d_in[3];
  const float* bk = (const float*)d_in[4];
  const float* Wv = (const float*)d_in[5];
  const float* bv = (const float*)d_in[6];
  // d_in[7] = Wr, d_in[8] = br : dead (softmax over size-1 axis == 1.0)
  const float* Wo = (const float*)d_in[9];
  const float* bo = (const float*)d_in[10];

  float* out = (float*)d_out;                       // [B,S,E] = 4194304 floats
  float* cw  = out + (size_t)M_TOTAL * E_;          // [B,S,1] = 4096 floats

  // workspace carve-up (bf16 elements)
  bf16* xb    = (bf16*)d_ws;                        // 4096*1024
  bf16* wqb   = xb  + (size_t)M_TOTAL * E_;         // 1024*1024 each
  bf16* wkb   = wqb + (size_t)E_ * E_;
  bf16* wvb   = wkb + (size_t)E_ * E_;
  bf16* wob   = wvb + (size_t)E_ * E_;
  bf16* qten  = wob + (size_t)E_ * E_;              // [B,H,S,DH]
  bf16* kten  = qten + (size_t)B_ * H_ * S_ * DH_;  // [B,H,S,DH]
  bf16* vTten = kten + (size_t)B_ * H_ * S_ * DH_;  // [B,H,DH,S]
  bf16* attnb = vTten + (size_t)B_ * H_ * S_ * DH_; // [B,S,E]

  // 1) fp32 -> bf16 conversions
  cvt_bf16_kernel<<<2048, 256, 0, stream>>>(x,  xb,  M_TOTAL * E_);
  cvt_bf16_kernel<<<1024, 256, 0, stream>>>(Wq, wqb, E_ * E_);
  cvt_bf16_kernel<<<1024, 256, 0, stream>>>(Wk, wkb, E_ * E_);
  cvt_bf16_kernel<<<1024, 256, 0, stream>>>(Wv, wvb, E_ * E_);
  cvt_bf16_kernel<<<1024, 256, 0, stream>>>(Wo, wob, E_ * E_);

  // 2) fused QKV projection (z selects matrix)
  {
    dim3 grid(M_TOTAL / 32, E_ / 256, 3);
    qkv_gemm_kernel<<<grid, 128, 0, stream>>>(xb, wqb, wkb, wvb, bq, bk, bv,
                                              qten, kten, vTten);
  }

  // 3) flash attention
  {
    dim3 grid(S_ / 64, B_ * H_);
    attn_kernel<<<grid, 128, 0, stream>>>(qten, kten, vTten, attnb);
  }

  // 4) output projection -> fp32 d_out
  {
    dim3 grid(M_TOTAL / 32, E_ / 256);
    out_gemm_kernel<<<grid, 128, 0, stream>>>(attnb, wob, bo, out);
  }

  // 5) cw == 1.0 identically
  fill_ones_kernel<<<(B_ * S_ + 255) / 256, 256, 0, stream>>>(cw, B_ * S_);
}
